// ConvLayer_17918603559109
// MI455X (gfx1250) — compile-verified, matrix operand order checked by reading, not run on hardware
//
#include <hip/hip_runtime.h>

// ---------------------------------------------------------------------------
// GCN conv layer: out = BN(relu(sum_k L_k @ (X @ W_k) + bias))
// N=8192 nodes, C_in=C_out=64.
// Memory-bound: 805 MB of L streamed from HBM @23.3TB/s (~35us floor) vs
// 25.8 GFLOP -> bf16 WMMA (f32 acc), NT b128 loads on L, L2-resident B.
// ---------------------------------------------------------------------------

typedef float  v4f   __attribute__((ext_vector_type(4)));
typedef float  v8f   __attribute__((ext_vector_type(8)));
typedef __bf16 v8bf  __attribute__((ext_vector_type(8)));
typedef __bf16 v16bf __attribute__((ext_vector_type(16)));

#define NNODES 8192
#define CDIM   64
#define KTOT   (3 * NNODES)   // concatenated K dimension

__device__ __forceinline__ unsigned short f2bf_bits(float f) {
    // round-to-nearest-even f32 -> bf16 (pure integer ops: guaranteed codegen)
    unsigned u = __builtin_bit_cast(unsigned, f);
    u += 0x7FFFu + ((u >> 16) & 1u);
    return (unsigned short)(u >> 16);
}
__device__ __forceinline__ __bf16 f2bf(float f) {
    unsigned short b = f2bf_bits(f);
    return __builtin_bit_cast(__bf16, b);
}

// ---------------------------------------------------------------------------
// Phase 1: Yt[c][kmat*8192 + m] = bf16( (X @ W_kmat)[m][c] )   (transposed,
// so phase-2 WMMA B-fragments are contiguous 16-byte loads per lane).
// ---------------------------------------------------------------------------
__global__ __launch_bounds__(128) void proj_kernel(
    const float* __restrict__ X,  const float* __restrict__ W0,
    const float* __restrict__ W1, const float* __restrict__ W2,
    unsigned short* __restrict__ Yt)
{
    __shared__ float Ws[64 * 64];
    __shared__ float Xs[128 * 65];          // +1 pad -> conflict-free banks
    const int tid  = threadIdx.x;
    const int kmat = blockIdx.y;
    const float* __restrict__ W = (kmat == 0) ? W0 : (kmat == 1) ? W1 : W2;

    for (int i = tid; i < 64 * 64; i += 128) Ws[i] = W[i];

    const int m = blockIdx.x * 128 + tid;
    const v4f* __restrict__ Xr = (const v4f*)(X + (size_t)m * 64);
    #pragma unroll
    for (int i = 0; i < 16; ++i) {
        v4f x = Xr[i];
        Xs[tid * 65 + 4 * i + 0] = x[0];
        Xs[tid * 65 + 4 * i + 1] = x[1];
        Xs[tid * 65 + 4 * i + 2] = x[2];
        Xs[tid * 65 + 4 * i + 3] = x[3];
    }
    __syncthreads();

    const size_t obase = (size_t)kmat * NNODES + m;
    for (int c = 0; c < 64; ++c) {
        float acc = 0.f;
        #pragma unroll 16
        for (int i = 0; i < 64; ++i)
            acc = fmaf(Xs[tid * 65 + i], Ws[i * 64 + c], acc);
        Yt[(size_t)c * KTOT + obase] = f2bf_bits(acc);   // coalesced across tid
    }
}

// ---------------------------------------------------------------------------
// Phase 2: P_kmat = L_kmat @ Y_kmat  via v_wmma_f32_16x16x32_bf16.
// Each wave: one 16-row strip x 64 cols (4 accumulators), K-loop of 32,
// unrolled 2x so next iteration's loads/converts pipeline under this
// iteration's WMMAs (avoids the per-iteration s_wait_loadcnt 0 drain).
// A (L rows) streamed f32 NT -> bf16; B (Yt) bf16 from L2.
// ISA A layout 16x32 bf16: lanes0-15 hold {K0-7,K16-23}, lanes16-31 {K8-15,K24-31}.
// ---------------------------------------------------------------------------
__global__ __launch_bounds__(128) void spmm_kernel(
    const float* __restrict__ L0, const float* __restrict__ L1,
    const float* __restrict__ L2, const unsigned short* __restrict__ Yt,
    float* __restrict__ P0, float* __restrict__ P1, float* __restrict__ P2)
{
    const int kmat = blockIdx.y;
    const float* __restrict__ L = (kmat == 0) ? L0 : (kmat == 1) ? L1 : L2;
    float*       __restrict__ P = (kmat == 0) ? P0 : (kmat == 1) ? P1 : P2;

    const int lane = threadIdx.x & 31;
    const int wave = threadIdx.x >> 5;
    const int half = lane >> 4;      // which K-half this lane holds
    const int mrow = lane & 15;      // A: row M ; B: col N ; D: col N
    const int m0   = (blockIdx.x * 4 + wave) * 16;

    const float* __restrict__ Abase =
        L + (size_t)(m0 + mrow) * NNODES + 8 * half;
    const unsigned short* __restrict__ Bbase =
        Yt + (size_t)kmat * NNODES + 8 * half;

    const v8f zero = {0.f, 0.f, 0.f, 0.f, 0.f, 0.f, 0.f, 0.f};
    v8f acc[4] = {zero, zero, zero, zero};

    #pragma unroll 2
    for (int ks = 0; ks < NNODES; ks += 32) {
        // ---- A fragment: 16 f32 per lane, streamed non-temporally ----
        const v4f* Ap = (const v4f*)(Abase + ks);
        v4f a0 = __builtin_nontemporal_load(Ap + 0);   // K = 8h + 0..3
        v4f a1 = __builtin_nontemporal_load(Ap + 1);   // K = 8h + 4..7
        v4f a2 = __builtin_nontemporal_load(Ap + 4);   // K = 16 + 8h + 0..3
        v4f a3 = __builtin_nontemporal_load(Ap + 5);   // K = 16 + 8h + 4..7
        __builtin_prefetch(Abase + ks + 128, 0, 0);    // global_prefetch_b8

        v16bf A;
        #pragma unroll
        for (int i = 0; i < 4; ++i) {
            A[i]      = f2bf(a0[i]);
            A[4 + i]  = f2bf(a1[i]);
            A[8 + i]  = f2bf(a2[i]);
            A[12 + i] = f2bf(a3[i]);
        }

        // ---- 4 N-tiles: B fragment = two contiguous 16B bf16 loads ----
        #pragma unroll
        for (int t = 0; t < 4; ++t) {
            const unsigned short* Bp =
                Bbase + (size_t)(t * 16 + mrow) * KTOT + ks;
            v8bf blo = *(const v8bf*)(Bp);        // K = 8h + 0..7
            v8bf bhi = *(const v8bf*)(Bp + 16);   // K = 16 + 8h + 0..7
            v16bf B;
            #pragma unroll
            for (int i = 0; i < 8; ++i) { B[i] = blo[i]; B[8 + i] = bhi[i]; }

            acc[t] = __builtin_amdgcn_wmma_f32_16x16x32_bf16(
                false, A, false, B, (short)0, acc[t], false, false);
        }
    }

    // D layout: element v of lane -> (M = m0 + v + 8*half, N = t*16 + mrow)
    #pragma unroll
    for (int t = 0; t < 4; ++t)
        #pragma unroll
        for (int v = 0; v < 8; ++v)
            P[(size_t)(m0 + v + 8 * half) * 64 + t * 16 + mrow] = acc[t][v];
}

// ---------------------------------------------------------------------------
// Phase 3a: per-channel BN stats over relu(P0+P1+P2+bias). One block / channel.
// ---------------------------------------------------------------------------
__global__ __launch_bounds__(256) void bn_stats_kernel(
    const float* __restrict__ P0, const float* __restrict__ P1,
    const float* __restrict__ P2, const float* __restrict__ bias,
    float* __restrict__ mean, float* __restrict__ rstd)
{
    __shared__ float ss[256], ss2[256];
    const int c = blockIdx.x, tid = threadIdx.x;
    const float b = bias[c];
    float s = 0.f, s2 = 0.f;
    for (int m = tid; m < NNODES; m += 256) {
        size_t i = (size_t)m * 64 + c;
        float v = fmaxf(P0[i] + P1[i] + P2[i] + b, 0.f);
        s += v; s2 += v * v;
    }
    ss[tid] = s; ss2[tid] = s2;
    __syncthreads();
    for (int off = 128; off > 0; off >>= 1) {
        if (tid < off) { ss[tid] += ss[tid + off]; ss2[tid] += ss2[tid + off]; }
        __syncthreads();
    }
    if (tid == 0) {
        float mu  = ss[0] * (1.f / NNODES);
        float var = ss2[0] * (1.f / NNODES) - mu * mu;
        mean[c] = mu;
        rstd[c] = rsqrtf(var + 1e-5f);
    }
}

// ---------------------------------------------------------------------------
// Phase 3b: out = (relu(P0+P1+P2+bias) - mean) * rstd * gamma + beta
// ---------------------------------------------------------------------------
__global__ __launch_bounds__(256) void bn_apply_kernel(
    const float* __restrict__ P0, const float* __restrict__ P1,
    const float* __restrict__ P2, const float* __restrict__ bias,
    const float* __restrict__ gamma, const float* __restrict__ beta,
    const float* __restrict__ mean, const float* __restrict__ rstd,
    float* __restrict__ out)
{
    const int idx = blockIdx.x * 256 + threadIdx.x;
    const int c = idx & 63;
    float v = fmaxf(P0[idx] + P1[idx] + P2[idx] + bias[c], 0.f);
    out[idx] = (v - mean[c]) * rstd[c] * gamma[c] + beta[c];
}

// ---------------------------------------------------------------------------
extern "C" void kernel_launch(void* const* d_in, const int* in_sizes, int n_in,
                              void* d_out, int out_size, void* d_ws, size_t ws_size,
                              hipStream_t stream)
{
    const float* X     = (const float*)d_in[0];
    const float* L0    = (const float*)d_in[1];
    const float* L1    = (const float*)d_in[2];
    const float* L2    = (const float*)d_in[3];
    const float* W0    = (const float*)d_in[4];
    const float* W1    = (const float*)d_in[5];
    const float* W2    = (const float*)d_in[6];
    const float* bias  = (const float*)d_in[7];
    const float* gamma = (const float*)d_in[8];
    const float* beta  = (const float*)d_in[9];
    float* out = (float*)d_out;

    // workspace layout (~9.5 MB): Yt bf16 [64 x 24576] | P0 P1 P2 f32 | stats
    char* ws = (char*)d_ws;
    unsigned short* Yt = (unsigned short*)ws;                 // 3,145,728 B
    float* P0   = (float*)(ws + (size_t)CDIM * KTOT * 2);
    float* P1   = P0 + (size_t)NNODES * CDIM;
    float* P2   = P1 + (size_t)NNODES * CDIM;
    float* mean = P2 + (size_t)NNODES * CDIM;
    float* rstd = mean + 64;

    proj_kernel<<<dim3(NNODES / 128, 3), dim3(128), 0, stream>>>(X, W0, W1, W2, Yt);
    spmm_kernel<<<dim3(NNODES / 64, 3), dim3(128), 0, stream>>>(L0, L1, L2, Yt, P0, P1, P2);
    bn_stats_kernel<<<dim3(64), dim3(256), 0, stream>>>(P0, P1, P2, bias, mean, rstd);
    bn_apply_kernel<<<dim3(NNODES * CDIM / 256), dim3(256), 0, stream>>>(
        P0, P1, P2, bias, gamma, beta, mean, rstd, out);
}
